// Transformer_41480794145180
// MI455X (gfx1250) — matrix operand order repaired and checked by reading, not compiled, not run on hardware
//
#include <hip/hip_runtime.h>
#include <hip/hip_bf16.h>
#include <math.h>

// ---------------------------------------------------------------------------
// Equivariant graph attention (NequIP/Equiformer-style) for MI455X (gfx1250).
//
//  * Per-edge einsum('huv,eu,ev->eh') hoisted to node level: A_i[n,h,v] =
//    sum_u q[n,u] W_i[h,u,v]; each edge then does cheap gathered dots.
//  * Edge MLPs ((E,8)@(8,64)->gelu->(E,64)@(64,128)) run on
//    v_wmma_f32_16x16x32_bf16 in 16-edge tiles per wave.
//  * B weights staged in LDS in the exact per-lane WMMA fragment layout so a
//    fragment is one aligned 32-byte LDS read (no per-element ds_load_u16).
//  * A fragments are two contiguous v8bf LDS reads + shufflevector.
//  * Nothing of size E x 128 is materialized; only expw (E x 4) hits memory.
//  * Node tables (41 MB) + node_f (5 MB) stay resident in the 192 MB L2.
// Workspace (floats): A0[N*128] A1[N*128] B2[N*384] B3[N*384]
//                     z[N*4] agg[N*256] expw[E*4]  => ~56.5 MB total.
// ---------------------------------------------------------------------------

#define N_NODES 10000
#define N_EDGES 320000
#define NB      8
#define HID     64
#define DIN     128

#define INV3        0.5773502691896258f
#define INV_FAN     0.015625f        /* 1/sqrt(4*32*32) */
#define INV_SQRT_NB 0.3535533905932738f
#define INV_SQRT_H  0.125f           /* 1/sqrt(64) */

typedef __attribute__((ext_vector_type(16))) __bf16 v16bf;
typedef __attribute__((ext_vector_type(8)))  __bf16 v8bf;
typedef __attribute__((ext_vector_type(8)))  float  v8f;

__device__ __forceinline__ float fast_tanh(float x) {
    float t = __expf(2.0f * x);
    return (t - 1.0f) / (t + 1.0f);
}
__device__ __forceinline__ float gelu_tanh(float x) {
    float x3 = x * x * x;
    return 0.5f * x * (1.0f + fast_tanh(0.7978845608028654f * (x + 0.044715f * x3)));
}

// ---------------------------------------------------------------------------
__global__ void zero_kernel(float* __restrict__ p, int n) {
    int i = blockIdx.x * 256 + threadIdx.x;
    if (i < n) p[i] = 0.0f;
}

// ---------------------------------------------------------------------------
// Node precompute: A0/A1[n,h,v] = sum_u xs[n,u] * W_dot[{0,1},h,u,v]
//                  B2/B3[n,h,v,d] = sum_u xv[n,u,d] * W_dot[{2,3},h,u,v]
// ---------------------------------------------------------------------------
__global__ void node_precompute_kernel(const float* __restrict__ node_f,
                                       const float* __restrict__ W_dot,
                                       float* __restrict__ ws) {
    __shared__ float nf[DIN];
    int n = blockIdx.x;
    int tid = threadIdx.x;                    // 128
    nf[tid] = node_f[(size_t)n * DIN + tid];
    __syncthreads();
    int h = tid >> 5, v = tid & 31;
    float a0 = 0.f, a1 = 0.f;
    float b20 = 0.f, b21 = 0.f, b22 = 0.f;
    float b30 = 0.f, b31 = 0.f, b32 = 0.f;
#pragma unroll 4
    for (int u = 0; u < 32; ++u) {
        int base = h * 1024 + u * 32 + v;
        float w0 = W_dot[base];
        float w1 = W_dot[4096 + base];
        float w2 = W_dot[8192 + base];
        float w3 = W_dot[12288 + base];
        float q  = nf[u];
        float q0 = nf[32 + u * 3 + 0];
        float q1 = nf[32 + u * 3 + 1];
        float q2 = nf[32 + u * 3 + 2];
        a0 += q * w0;  a1 += q * w1;
        b20 += q0 * w2; b21 += q1 * w2; b22 += q2 * w2;
        b30 += q0 * w3; b31 += q1 * w3; b32 += q2 * w3;
    }
    float* A0 = ws;
    float* A1 = ws + (size_t)N_NODES * 128;
    float* B2 = ws + (size_t)N_NODES * 256;
    float* B3 = ws + (size_t)N_NODES * 640;
    A0[(size_t)n * 128 + h * 32 + v] = a0;
    A1[(size_t)n * 128 + h * 32 + v] = a1;
    size_t bo = (size_t)n * 384 + h * 96 + v * 3;
    B2[bo + 0] = b20; B2[bo + 1] = b21; B2[bo + 2] = b22;
    B3[bo + 0] = b30; B3[bo + 1] = b31; B3[bo + 2] = b32;
}

// ---------------------------------------------------------------------------
// Per-wave fused MLP for a 16-edge tile.
//   s_w1t : [HID][NB]  layer-1 weights transposed (contiguous K per feature)
//   s_w2f : [8 tiles][2 chunks][32 lanes][16 bf16]  fragment-packed layer-2
//   s_h   : per-wave [16][HID] bf16 activations
//   s_o   : per-wave [16][128] f32 MLP output
// A layout (16-bit A 16x32): lanes 0-15 row=lane, elems 0..7 -> K 0..7,
// elems 8..15 -> K 16..23; lanes 16-31 same rows, K 8..15 / 24..31.
// ---------------------------------------------------------------------------
__device__ __forceinline__ void mlp16(int lane, int e,
                                      const float* __restrict__ sattr,
                                      const float* s_w1t, const float* s_b1,
                                      const __bf16* s_w2f,
                                      __bf16* s_h, float* s_o) {
    int el = lane & 15;     // edge within tile
    int fh = lane >> 4;     // feature half / register half

    float4 sa0 = *(const float4*)(sattr + (size_t)e * NB);
    float4 sa1 = *(const float4*)(sattr + (size_t)e * NB + 4);

    __bf16 hv[32];
#pragma unroll
    for (int f0 = 0; f0 < 32; ++f0) {
        int f = fh * 32 + f0;
        float4 wa = *(const float4*)&s_w1t[f * 8];
        float4 wb = *(const float4*)&s_w1t[f * 8 + 4];
        float acc = sa0.x * wa.x + sa0.y * wa.y + sa0.z * wa.z + sa0.w * wa.w
                  + sa1.x * wb.x + sa1.y * wb.y + sa1.z * wb.z + sa1.w * wb.w;
        acc = acc * INV_SQRT_NB + s_b1[f];
        hv[f0] = (__bf16)gelu_tanh(acc);
    }
    v8bf* hp = (v8bf*)(s_h + el * HID + fh * 32);
#pragma unroll
    for (int q = 0; q < 4; ++q) {
        v8bf t;
#pragma unroll
        for (int j = 0; j < 8; ++j) t[j] = hv[q * 8 + j];
        hp[q] = t;
    }
    __syncthreads();   // activations cross lanes within the wave

    v8f acc[8];
#pragma unroll
    for (int t = 0; t < 8; ++t)
#pragma unroll
        for (int j = 0; j < 8; ++j) acc[t][j] = 0.0f;

#pragma unroll
    for (int c = 0; c < 2; ++c) {          // K chunks of 32
        const v8bf* ap = (const v8bf*)(s_h + el * HID + c * 32 + fh * 8);
        v8bf alo = ap[0];                  // K base + {0..7}   (this half)
        v8bf ahi = ap[2];                  // K base + 16 + {0..7}
        v16bf a = __builtin_shufflevector(alo, ahi,
                    0, 1, 2, 3, 4, 5, 6, 7, 8, 9, 10, 11, 12, 13, 14, 15);
#pragma unroll
        for (int t = 0; t < 8; ++t) {      // 8 N-tiles of 16 -> 128 outputs
            v16bf b = *(const v16bf*)(s_w2f + (((t * 2 + c) * 32 + lane) << 4));
            acc[t] = __builtin_amdgcn_wmma_f32_16x16x32_bf16(
                false, a, false, b, (short)0, acc[t], false, false);
        }
    }
    // D layout: lane col = lane&15; VGPR j -> row j (lanes<16) / 8+j (lanes>=16)
#pragma unroll
    for (int t = 0; t < 8; ++t)
#pragma unroll
        for (int j = 0; j < 8; ++j) {
            int m = fh * 8 + j;
            int n = t * 16 + el;
            s_o[m * 128 + n] = acc[t][j] * INV_SQRT_H;
        }
    __syncthreads();
}

// Shared-weight staging used by both edge kernels (tid in [0,64)).
__device__ __forceinline__ void stage_weights(int tid,
                                              const float* __restrict__ w1,
                                              const float* __restrict__ b1,
                                              const float* __restrict__ w2,
                                              float* s_w1t, float* s_b1,
                                              __bf16* s_w2f) {
    for (int i = tid; i < NB * HID; i += 64) {      // transpose w1 -> [f][k]
        int f = i >> 3, k = i & 7;
        s_w1t[i] = w1[k * HID + f];
    }
    if (tid < HID) s_b1[tid] = b1[tid];
    for (int i = tid; i < 16 * 32 * 16; i += 64) {  // fragment-pack w2
        int j    = i & 15;
        int frag = i >> 4;
        int ln   = frag & 31;
        int tc   = frag >> 5;        // t*2 + c
        int c    = tc & 1, t = tc >> 1;
        int half = ln >> 4, col = ln & 15;
        int k = c * 32 + half * 8 + (j & 7) + (j >> 3) * 16;
        int n = t * 16 + col;
        s_w2f[i] = (__bf16)w2[k * 128 + n];
    }
}

// ---------------------------------------------------------------------------
// Pass 1: wk = MLP_k, tensor-product keys, logits d[e,h], expw, atomic z.
// Block = 64 threads = 2 waves; each wave owns 16 edges; 2 lanes per edge.
// ---------------------------------------------------------------------------
__global__ void __launch_bounds__(64)
edge_pass1_kernel(const int* __restrict__ edge_src, const int* __restrict__ edge_dst,
                  const float* __restrict__ sattr, const float* __restrict__ eattr,
                  const float* __restrict__ cutoff, const float* __restrict__ node_f,
                  const float* __restrict__ wk1, const float* __restrict__ bk1,
                  const float* __restrict__ wk2, float* __restrict__ ws) {
    __shared__ float  s_w1t[NB * HID];
    __shared__ float  s_b1[HID];
    __shared__ __bf16 s_w2f[16 * 32 * 16];
    __shared__ __bf16 s_h[2][16 * HID];
    __shared__ float  s_o[2][16 * 128];

    int tid = threadIdx.x, wave = tid >> 5, lane = tid & 31;
    stage_weights(tid, wk1, bk1, wk2, s_w1t, s_b1, s_w2f);
    __syncthreads();

    int eBase = blockIdx.x * 32 + wave * 16;
    int el = lane & 15, sub = lane >> 4;
    int e = eBase + el;                       // E % 32 == 0: no tail
    mlp16(lane, e, sattr, s_w1t, s_b1, s_w2f, s_h[wave], s_o[wave]);

    const float* A0 = ws;
    const float* A1 = ws + (size_t)N_NODES * 128;
    const float* B2 = ws + (size_t)N_NODES * 256;
    const float* B3 = ws + (size_t)N_NODES * 640;
    float* zbuf = ws + (size_t)N_NODES * 1024;
    float* expw = ws + (size_t)N_NODES * 1284;

    int src = edge_src[e], dst = edge_dst[e];
    float4 ea = *(const float4*)(eattr + (size_t)e * 4);
    float ys = ea.x, yv0 = ea.y, yv1 = ea.z, yv2 = ea.w;
    const float* nf  = node_f + (size_t)src * DIN;
    const float* wkr = s_o[wave] + el * 128;
    const float* A0p = A0 + (size_t)dst * 128;
    const float* A1p = A1 + (size_t)dst * 128;
    const float* B2p = B2 + (size_t)dst * 384;
    const float* B3p = B3 + (size_t)dst * 384;
    __builtin_prefetch(nf, 0, 0);
    __builtin_prefetch(A0p, 0, 0);

    float dh[4] = {0.f, 0.f, 0.f, 0.f};
#pragma unroll
    for (int q = 0; q < 4; ++q) {
        int vb = sub * 16 + q * 4;
        float g0[4], g1[4], g2[4], g3[4];
        *(float4*)g0 = *(const float4*)&wkr[vb];
        *(float4*)g1 = *(const float4*)&wkr[32 + vb];
        *(float4*)g2 = *(const float4*)&wkr[64 + vb];
        *(float4*)g3 = *(const float4*)&wkr[96 + vb];
#pragma unroll
        for (int r = 0; r < 4; ++r) {
            int v = vb + r;
            float sxs = nf[v];
            float sv0 = nf[32 + v * 3 + 0];
            float sv1 = nf[32 + v * 3 + 1];
            float sv2 = nf[32 + v * 3 + 2];
            float k0 = g0[r] * sxs * ys;
            float k1 = g1[r] * (sv0 * yv0 + sv1 * yv1 + sv2 * yv2) * INV3;
            float kp = g2[r] * sxs;          // kv0[v][d] = kp * yv[d]
            float kq = g3[r] * ys;           // kv1[v][d] = kq * sxv[v][d]
#pragma unroll
            for (int h = 0; h < 4; ++h) {
                float t = A0p[h * 32 + v] * k0 + A1p[h * 32 + v] * k1;
                int bo = h * 96 + v * 3;
                float b2 = B2p[bo] * yv0 + B2p[bo + 1] * yv1 + B2p[bo + 2] * yv2;
                float b3 = B3p[bo] * sv0 + B3p[bo + 1] * sv1 + B3p[bo + 2] * sv2;
                dh[h] += t + INV3 * (kp * b2 + kq * b3);
            }
        }
    }
#pragma unroll
    for (int h = 0; h < 4; ++h) dh[h] += __shfl_xor(dh[h], 16);
    if (sub == 0) {
        float cw = cutoff[e];
#pragma unroll
        for (int h = 0; h < 4; ++h) {
            float w = cw * __expf(dh[h] * INV_FAN);
            expw[(size_t)e * 4 + h] = w;
            atomicAdd(&zbuf[(size_t)dst * 4 + h], w);
        }
    }
}

// ---------------------------------------------------------------------------
// Pass 2: wv = MLP_v, values scaled by sqrt(relu(alpha)), scatter-add to agg.
// ---------------------------------------------------------------------------
__global__ void __launch_bounds__(64)
edge_pass2_kernel(const int* __restrict__ edge_src, const int* __restrict__ edge_dst,
                  const float* __restrict__ sattr, const float* __restrict__ eattr,
                  const float* __restrict__ node_f,
                  const float* __restrict__ wv1, const float* __restrict__ bv1,
                  const float* __restrict__ wv2, float* __restrict__ ws) {
    __shared__ float  s_w1t[NB * HID];
    __shared__ float  s_b1[HID];
    __shared__ __bf16 s_w2f[16 * 32 * 16];
    __shared__ __bf16 s_h[2][16 * HID];
    __shared__ float  s_o[2][16 * 128];

    int tid = threadIdx.x, wave = tid >> 5, lane = tid & 31;
    stage_weights(tid, wv1, bv1, wv2, s_w1t, s_b1, s_w2f);
    __syncthreads();

    int eBase = blockIdx.x * 32 + wave * 16;
    int el = lane & 15, sub = lane >> 4;
    int e = eBase + el;
    mlp16(lane, e, sattr, s_w1t, s_b1, s_w2f, s_h[wave], s_o[wave]);

    const float* zbuf = ws + (size_t)N_NODES * 1024;
    float*       agg  = ws + (size_t)N_NODES * 1028;
    const float* expw = ws + (size_t)N_NODES * 1284;

    int src = edge_src[e], dst = edge_dst[e];
    float4 ea = *(const float4*)(eattr + (size_t)e * 4);
    float ys = ea.x, yv0 = ea.y, yv1 = ea.z, yv2 = ea.w;
    const float* nf  = node_f + (size_t)src * DIN;
    const float* wvr = s_o[wave] + el * 128;
    float* ag = agg + (size_t)dst * 256;
    __builtin_prefetch(nf, 0, 0);

    float sa[4];
#pragma unroll
    for (int h = 0; h < 4; ++h) {
        float z = zbuf[(size_t)dst * 4 + h];
        z = (z == 0.0f) ? 1.0f : z;
        float al = expw[(size_t)e * 4 + h] / z;
        sa[h] = sqrtf(fmaxf(al, 0.0f));
    }
#pragma unroll
    for (int q = 0; q < 4; ++q) {
        int vb = sub * 16 + q * 4;
        float g0[4], g1[4], g2[4], g3[4];
        *(float4*)g0 = *(const float4*)&wvr[vb];
        *(float4*)g1 = *(const float4*)&wvr[32 + vb];
        *(float4*)g2 = *(const float4*)&wvr[64 + vb];
        *(float4*)g3 = *(const float4*)&wvr[96 + vb];
#pragma unroll
        for (int r = 0; r < 4; ++r) {
            int v = vb + r;
            float s = sa[v >> 3];             // head = v / 8
            float sxs = nf[v];
            float sv0 = nf[32 + v * 3 + 0];
            float sv1 = nf[32 + v * 3 + 1];
            float sv2 = nf[32 + v * 3 + 2];
            float dotv = sv0 * yv0 + sv1 * yv1 + sv2 * yv2;
            float e0 = g0[r] * sxs * ys * s;
            float e1 = g1[r] * dotv * INV3 * s;
            float c2 = g2[r] * sxs * s;       // e2[v][d] = c2*yv[d]
            float c3 = g3[r] * ys  * s;       // e3[v][d] = c3*sxv[v][d]
            atomicAdd(&ag[v],            e0);
            atomicAdd(&ag[32 + v],       e1);
            atomicAdd(&ag[64 + v * 3 + 0],  c2 * yv0);
            atomicAdd(&ag[64 + v * 3 + 1],  c2 * yv1);
            atomicAdd(&ag[64 + v * 3 + 2],  c2 * yv2);
            atomicAdd(&ag[160 + v * 3 + 0], c3 * sv0);
            atomicAdd(&ag[160 + v * 3 + 1], c3 * sv1);
            atomicAdd(&ag[160 + v * 3 + 2], c3 * sv2);
        }
    }
}

// ---------------------------------------------------------------------------
// Node output: out_s = s @ W_lin_s / 8 ; out_v[w,d] = sum_u v[u,d] W_lin_v[u,w] / 8
// ---------------------------------------------------------------------------
__global__ void node_out_kernel(const float* __restrict__ Ws,
                                const float* __restrict__ Wv,
                                const float* __restrict__ ws,
                                float* __restrict__ out) {
    __shared__ float row[256];
    int n = blockIdx.x, t = threadIdx.x;      // 128
    const float* agg = ws + (size_t)N_NODES * 1028 + (size_t)n * 256;
    row[t] = agg[t];
    row[128 + t] = agg[128 + t];
    __syncthreads();
    if (t < 32) {
        float s = 0.0f;
#pragma unroll 8
        for (int u = 0; u < 64; ++u) s += row[u] * Ws[u * 32 + t];
        out[(size_t)n * 128 + t] = s * 0.125f;
    } else {
        int p = t - 32;                        // 0..95
        int w = p / 3, d = p - w * 3;
        float s = 0.0f;
#pragma unroll 8
        for (int u = 0; u < 64; ++u) s += row[64 + u * 3 + d] * Wv[u * 32 + w];
        out[(size_t)n * 128 + 32 + p] = s * 0.125f;
    }
}

// ---------------------------------------------------------------------------
extern "C" void kernel_launch(void* const* d_in, const int* in_sizes, int n_in,
                              void* d_out, int out_size, void* d_ws, size_t ws_size,
                              hipStream_t stream) {
    const int*   edge_src = (const int*)d_in[0];
    const int*   edge_dst = (const int*)d_in[1];
    const float* sattr    = (const float*)d_in[2];
    const float* eattr    = (const float*)d_in[3];
    const float* cutoff   = (const float*)d_in[4];
    const float* node_f   = (const float*)d_in[5];
    const float* wk1      = (const float*)d_in[6];
    const float* bk1      = (const float*)d_in[7];
    const float* wk2      = (const float*)d_in[8];
    const float* wv1      = (const float*)d_in[9];
    const float* bv1      = (const float*)d_in[10];
    const float* wv2      = (const float*)d_in[11];
    const float* W_dot    = (const float*)d_in[12];
    const float* W_lin_s  = (const float*)d_in[13];
    const float* W_lin_v  = (const float*)d_in[14];
    float* ws  = (float*)d_ws;     // needs (N*1284 + E*4)*4 bytes ~= 56.5 MB
    float* out = (float*)d_out;

    int zn = N_NODES * 260;        // z (N*4) + agg (N*256)
    zero_kernel<<<(zn + 255) / 256, 256, 0, stream>>>(ws + (size_t)N_NODES * 1024, zn);
    node_precompute_kernel<<<N_NODES, 128, 0, stream>>>(node_f, W_dot, ws);
    edge_pass1_kernel<<<N_EDGES / 32, 64, 0, stream>>>(edge_src, edge_dst, sattr, eattr,
                                                       cutoff, node_f, wk1, bk1, wk2, ws);
    edge_pass2_kernel<<<N_EDGES / 32, 64, 0, stream>>>(edge_src, edge_dst, sattr, eattr,
                                                       node_f, wv1, bv1, wv2, ws);
    node_out_kernel<<<N_NODES, 128, 0, stream>>>(W_lin_s, W_lin_v, ws, out);
}
